// PersonalizedSimpleAttention_13846974562350
// MI455X (gfx1250) — matrix operand order repaired and checked by reading, not compiled, not run on hardware
//
#include <hip/hip_runtime.h>
#include <math.h>

// ---------------------------------------------------------------------------
// PersonalizedSimpleAttention on gfx1250 (MI455X), bf16 WMMA pipeline.
//   B=64, T=1024, EMB=128, KH=VH=256, NP=32
// Pass 0: one-shot f32->bf16 conversion of x + all GEMM weights (bandwidth
//         bound; removes per-wave conversion VALU from the hot kernels).
// Pass 1: QKV projections + routed per-person linear (all-WMMA).
// Pass 2: flash attention with online softmax (all-WMMA + shuffles).
// ---------------------------------------------------------------------------

#define BB   64
#define TT   1024
#define EMB  128
#define KH   256

typedef __attribute__((ext_vector_type(16))) __bf16 v16bf;
typedef __attribute__((ext_vector_type(8)))  __bf16 v8bf;
typedef __attribute__((ext_vector_type(8)))  float  v8f;

__device__ inline unsigned short f2bfu(float f) {
    unsigned int u = __float_as_uint(f);
    u += 0x7FFFu + ((u >> 16) & 1u);        // round-to-nearest-even
    return (unsigned short)(u >> 16);
}
__device__ inline __bf16 u2bf(unsigned short h) { return __builtin_bit_cast(__bf16, h); }

__device__ inline v8f wmma_bf16(v16bf a, v16bf b, v8f c) {
    return __builtin_amdgcn_wmma_f32_16x16x32_bf16(false, a, false, b, (short)0, c, false, false);
}

// A-fragment (16x32 bf16): lane row = lane&15; element e -> k = kbase +
// hi*16 + e. Per lane that is one contiguous 16-element run.
__device__ inline v16bf afrag(const unsigned short* rowp, int kbase, int hi) {
    v8bf lo = *(const v8bf*)(rowp + kbase + hi * 16);
    v8bf hh = *(const v8bf*)(rowp + kbase + hi * 16 + 8);
    return __builtin_shufflevector(lo, hh, 0, 1, 2, 3, 4, 5, 6, 7,
                                   8, 9, 10, 11, 12, 13, 14, 15);
}

// ===========================================================================
// Pass 0: bulk f32 -> bf16 (4 elements / thread, 8-byte packed stores).
// ===========================================================================
__global__ __launch_bounds__(256)
void cvt_bf16_kernel(const float* __restrict__ src,
                     unsigned short* __restrict__ dst, int n)
{
    int i = (blockIdx.x * 256 + threadIdx.x) * 4;
    if (i < n) {
        float4 f = *(const float4*)(src + i);
        unsigned long long p =
            (unsigned long long)f2bfu(f.x) |
            ((unsigned long long)f2bfu(f.y) << 16) |
            ((unsigned long long)f2bfu(f.z) << 32) |
            ((unsigned long long)f2bfu(f.w) << 48);
        *(unsigned long long*)(dst + i) = p;
    }
}

// ===========================================================================
// Pass 1: shared projections + routed per-person linear.
//   q_ws[b][t][d] (bf16, pre-scaled by 1/sqrt(KH)), k_ws[b][t][d] (bf16),
//   vT_ws[b][d][t] (bf16 transposed -> contiguous PV B-fragments).
// One wave per 16-row tile of T; 4 waves / block.
// ===========================================================================
__global__ __launch_bounds__(128)
void qkv_route_kernel(const unsigned short* __restrict__ xbf,
                      const int*   __restrict__ pidx,
                      const unsigned short* __restrict__ wk,
                      const unsigned short* __restrict__ wq,
                      const unsigned short* __restrict__ wv,
                      const unsigned short* __restrict__ pkw,
                      const float* __restrict__ PKb,
                      const unsigned short* __restrict__ pqw,
                      const float* __restrict__ PQb,
                      unsigned short* __restrict__ q_ws,
                      unsigned short* __restrict__ k_ws,
                      unsigned short* __restrict__ vT_ws)
{
    __shared__ alignas(32) unsigned short kq_lds[4][2][16 * KH];

    const int lane = threadIdx.x & 31;
    const int w    = threadIdx.x >> 5;
    const int ln   = lane & 15;
    const int hi   = lane >> 4;
    const int tile = blockIdx.x * 4 + w;
    const int b    = tile >> 6;
    const int t0   = (tile & 63) << 4;
    const int p    = pidx[b];

    // ---- x A-fragments (4 k-chunks over EMB=128) ---------------------------
    v16bf xa[4];
    {
        const unsigned short* xr = xbf + (size_t)(b * TT + t0 + ln) * EMB;
#pragma unroll
        for (int kc = 0; kc < 4; ++kc) xa[kc] = afrag(xr, kc * 32, hi);
    }

    // ---- V projection -> vT_ws (transposed) --------------------------------
#pragma unroll 1
    for (int n = 0; n < 16; ++n) {
        const unsigned short* wr = wv + (size_t)(n * 16 + ln) * EMB + hi * 16;
        v8f acc = {0.f, 0.f, 0.f, 0.f, 0.f, 0.f, 0.f, 0.f};
#pragma unroll
        for (int kc = 0; kc < 4; ++kc)
            acc = wmma_bf16(xa[kc], *(const v16bf*)(wr + kc * 32), acc);
#pragma unroll
        for (int r = 0; r < 8; ++r)
            vT_ws[((size_t)b * KH + n * 16 + ln) * TT + t0 + r + hi * 8] = f2bfu(acc[r]);
    }

    // ---- K0 / Q0 projections -> LDS (bf16) ---------------------------------
#pragma unroll 1
    for (int n = 0; n < 16; ++n) {
        const unsigned short* wkr = wk + (size_t)(n * 16 + ln) * EMB + hi * 16;
        const unsigned short* wqr = wq + (size_t)(n * 16 + ln) * EMB + hi * 16;
        v8f ak = {0.f, 0.f, 0.f, 0.f, 0.f, 0.f, 0.f, 0.f};
        v8f aq = {0.f, 0.f, 0.f, 0.f, 0.f, 0.f, 0.f, 0.f};
#pragma unroll
        for (int kc = 0; kc < 4; ++kc) {
            ak = wmma_bf16(xa[kc], *(const v16bf*)(wkr + kc * 32), ak);
            aq = wmma_bf16(xa[kc], *(const v16bf*)(wqr + kc * 32), aq);
        }
#pragma unroll
        for (int r = 0; r < 8; ++r) {
            int idx = (r + hi * 8) * KH + n * 16 + ln;
            kq_lds[w][0][idx] = f2bfu(ak[r]);
            kq_lds[w][1][idx] = f2bfu(aq[r]);
        }
    }
    // in-wave LDS store -> load hazard (CDNA5 split counters)
    asm volatile("s_wait_dscnt 0x0" ::: "memory");

    // ---- routed linears: keys/queries = K0/Q0 @ P*_W[p]^T + P*_b[p] --------
    const unsigned short* k0row = &kq_lds[w][0][ln * KH];
    const unsigned short* q0row = &kq_lds[w][1][ln * KH];
#pragma unroll 1
    for (int n = 0; n < 16; ++n) {
        float bk = PKb[(size_t)p * KH + n * 16 + ln];
        float bq = PQb[(size_t)p * KH + n * 16 + ln];
        v8f ack, acq;
#pragma unroll
        for (int r = 0; r < 8; ++r) { ack[r] = bk; acq[r] = bq; }
        const unsigned short* wkp = pkw + ((size_t)p * KH + n * 16 + ln) * KH + hi * 16;
        const unsigned short* wqp = pqw + ((size_t)p * KH + n * 16 + ln) * KH + hi * 16;
#pragma unroll
        for (int kc = 0; kc < 8; ++kc) {
            v16bf a0 = afrag(k0row, kc * 32, hi);
            v16bf a1 = afrag(q0row, kc * 32, hi);
            ack = wmma_bf16(a0, *(const v16bf*)(wkp + kc * 32), ack);
            acq = wmma_bf16(a1, *(const v16bf*)(wqp + kc * 32), acq);
        }
#pragma unroll
        for (int r = 0; r < 8; ++r) {
            size_t off = ((size_t)(b * TT + t0 + r + hi * 8)) * KH + n * 16 + ln;
            k_ws[off] = f2bfu(ack[r]);
            q_ws[off] = f2bfu(acq[r] * 0.0625f);   // 1/sqrt(KH)
        }
    }
}

// ===========================================================================
// Pass 2: flash attention. One wave per 16-query tile, key chunks of 32,
// online softmax, O[16,256] accumulated in f32 registers.
// ===========================================================================
__global__ __launch_bounds__(128)
void attn_kernel(const unsigned short* __restrict__ q_ws,
                 const unsigned short* __restrict__ k_ws,
                 const unsigned short* __restrict__ vT_ws,
                 const unsigned char*  __restrict__ mask,
                 float* __restrict__ out)
{
    __shared__ alignas(32) unsigned short p_lds[4][16 * 32];

    const int lane = threadIdx.x & 31;
    const int w    = threadIdx.x >> 5;
    const int ln   = lane & 15;
    const int hi   = lane >> 4;
    const int tile = blockIdx.x * 4 + w;
    const int b    = tile >> 6;
    const int t0   = (tile & 63) << 4;

    v8f o[16];
#pragma unroll
    for (int n = 0; n < 16; ++n)
#pragma unroll
        for (int r = 0; r < 8; ++r) o[n][r] = 0.f;

    float m_run[8], l_run[8];
#pragma unroll
    for (int r = 0; r < 8; ++r) { m_run[r] = -1.0e30f; l_run[r] = 0.f; }

    // ---- preload Q A-fragments (whole d=256) -------------------------------
    v16bf qa[8];
    {
        const unsigned short* qr = q_ws + ((size_t)(b * TT + t0 + ln)) * KH;
#pragma unroll
        for (int kc = 0; kc < 8; ++kc) qa[kc] = afrag(qr, kc * 32, hi);
    }

    const unsigned short* vbase = vT_ws + (size_t)b * KH * TT;

#pragma unroll 1
    for (int s0 = 0; s0 < TT; s0 += 32) {
        // ---- S = Q K^T for two 16-key tiles --------------------------------
        v8f st0, st1;
#pragma unroll
        for (int r = 0; r < 8; ++r) { st0[r] = 0.f; st1[r] = 0.f; }
        const unsigned short* kr0 = k_ws + ((size_t)(b * TT + s0 + ln)) * KH + hi * 16;
        const unsigned short* kr1 = kr0 + (size_t)16 * KH;
#pragma unroll
        for (int kc = 0; kc < 8; ++kc) {
            st0 = wmma_bf16(qa[kc], *(const v16bf*)(kr0 + kc * 32), st0);
            st1 = wmma_bf16(qa[kc], *(const v16bf*)(kr1 + kc * 32), st1);
        }

        const bool mok0 = mask[(size_t)b * TT + s0 + ln] != 0;
        const bool mok1 = mask[(size_t)b * TT + s0 + 16 + ln] != 0;

        // ---- online softmax (rows r + 8*hi; 16 cols across the half-wave) --
        float alpha[8];
#pragma unroll
        for (int r = 0; r < 8; ++r) {
            float a0 = mok0 ? st0[r] : -1.0e30f;
            float a1 = mok1 ? st1[r] : -1.0e30f;
            float mx = fmaxf(a0, a1);
            mx = fmaxf(mx, __shfl_xor(mx, 1, 32));
            mx = fmaxf(mx, __shfl_xor(mx, 2, 32));
            mx = fmaxf(mx, __shfl_xor(mx, 4, 32));
            mx = fmaxf(mx, __shfl_xor(mx, 8, 32));
            float mnew = fmaxf(m_run[r], mx);
            float al   = __expf(m_run[r] - mnew);
            m_run[r]   = mnew;
            float p0 = __expf(a0 - mnew);
            float p1 = __expf(a1 - mnew);
            p_lds[w][(r + hi * 8) * 32 + ln]      = f2bfu(p0);
            p_lds[w][(r + hi * 8) * 32 + 16 + ln] = f2bfu(p1);
            float rs = p0 + p1;
            rs += __shfl_xor(rs, 1, 32);
            rs += __shfl_xor(rs, 2, 32);
            rs += __shfl_xor(rs, 4, 32);
            rs += __shfl_xor(rs, 8, 32);
            l_run[r] = al * l_run[r] + rs;
            alpha[r] = al;
        }
        asm volatile("s_wait_dscnt 0x0" ::: "memory");

        // ---- P A-fragment (D-layout -> A-layout via LDS) -------------------
        v16bf pa = afrag(&p_lds[w][ln * 32], 0, hi);

        // ---- O = alpha*O + P @ V (C pre-scale folds into WMMA) -------------
#pragma unroll
        for (int n = 0; n < 16; ++n) {
            v8f c = o[n];
#pragma unroll
            for (int r = 0; r < 8; ++r) c[r] *= alpha[r];
            v16bf bv = *(const v16bf*)(vbase + (size_t)(n * 16 + ln) * TT + s0 + hi * 16);
            o[n] = wmma_bf16(pa, bv, c);
        }
    }

    // ---- normalize and store -----------------------------------------------
    float inv[8];
#pragma unroll
    for (int r = 0; r < 8; ++r) inv[r] = 1.0f / l_run[r];
    float* orow = out + ((size_t)(b * TT + t0)) * KH;
#pragma unroll
    for (int n = 0; n < 16; ++n)
#pragma unroll
        for (int r = 0; r < 8; ++r)
            orow[(size_t)(r + hi * 8) * KH + n * 16 + ln] = o[n][r] * inv[r];
}

// ===========================================================================
extern "C" void kernel_launch(void* const* d_in, const int* in_sizes, int n_in,
                              void* d_out, int out_size, void* d_ws, size_t ws_size,
                              hipStream_t stream)
{
    const float*         x    = (const float*)d_in[0];
    const unsigned char* mask = (const unsigned char*)d_in[1]; // bool, 1 byte
    const int*           pidx = (const int*)d_in[2];
    const float*         Wk   = (const float*)d_in[3];
    const float*         Wq   = (const float*)d_in[4];
    const float*         Wv   = (const float*)d_in[5];
    const float*         PKW  = (const float*)d_in[6];
    const float*         PKb  = (const float*)d_in[7];
    const float*         PQW  = (const float*)d_in[8];
    const float*         PQb  = (const float*)d_in[9];
    float*               out  = (float*)d_out;

    // --- workspace layout (bf16 elements) -----------------------------------
    unsigned short* q_ws  = (unsigned short*)d_ws;
    unsigned short* k_ws  = q_ws  + (size_t)BB * TT * KH;
    unsigned short* v_ws  = k_ws  + (size_t)BB * TT * KH;
    unsigned short* xbf   = v_ws  + (size_t)BB * TT * KH;
    unsigned short* wkbf  = xbf   + (size_t)BB * TT * EMB;
    unsigned short* wqbf  = wkbf  + (size_t)KH * EMB;
    unsigned short* wvbf  = wqbf  + (size_t)KH * EMB;
    unsigned short* pkwbf = wvbf  + (size_t)KH * EMB;
    unsigned short* pqwbf = pkwbf + (size_t)32 * KH * KH;

    // --- Pass 0: bulk conversions -------------------------------------------
    const int NX  = BB * TT * EMB;
    const int NW  = KH * EMB;
    const int NPW = 32 * KH * KH;
    cvt_bf16_kernel<<<(NX  / 4 + 255) / 256, 256, 0, stream>>>(x,   xbf,   NX);
    cvt_bf16_kernel<<<(NW  / 4 + 255) / 256, 256, 0, stream>>>(Wk,  wkbf,  NW);
    cvt_bf16_kernel<<<(NW  / 4 + 255) / 256, 256, 0, stream>>>(Wq,  wqbf,  NW);
    cvt_bf16_kernel<<<(NW  / 4 + 255) / 256, 256, 0, stream>>>(Wv,  wvbf,  NW);
    cvt_bf16_kernel<<<(NPW / 4 + 255) / 256, 256, 0, stream>>>(PKW, pkwbf, NPW);
    cvt_bf16_kernel<<<(NPW / 4 + 255) / 256, 256, 0, stream>>>(PQW, pqwbf, NPW);

    const int nblocks = (BB * TT / 16) / 4;   // 1024 blocks, 4 waves each

    qkv_route_kernel<<<nblocks, 128, 0, stream>>>(xbf, pidx, wkbf, wqbf, wvbf,
                                                  pkwbf, PKb, pqwbf, PQb,
                                                  q_ws, k_ws, v_ws);
    attn_kernel<<<nblocks, 128, 0, stream>>>(q_ws, k_ws, v_ws, mask, out);
}
